// RNN_34291018891897
// MI455X (gfx1250) — compile-verified, matrix operand order checked by reading, not run on hardware
//
#include <hip/hip_runtime.h>

// ---------------------------------------------------------------------------
// LSTM on MI455X (gfx1250).  Sizes fixed by the reference: T=4096, B=32, H=256.
// ---------------------------------------------------------------------------
typedef __bf16  bf16_t;
typedef bf16_t  v16bf __attribute__((ext_vector_type(16)));
typedef float   v8f   __attribute__((ext_vector_type(8)));
typedef unsigned int v4u __attribute__((ext_vector_type(4)));
typedef int     v8i   __attribute__((ext_vector_type(8)));
typedef int     v4i   __attribute__((ext_vector_type(4)));

#define T_STEPS 4096
#define BATCH   32
#define HID     256
#define G4H     1024          // 4*H
#define TBROWS  (T_STEPS * BATCH)
#define NWG     8             // persistent workgroups in the scan phase

__device__ __forceinline__ v8f wmma_bf16(v16bf a, v16bf b, v8f c) {
  // D = A(16x32 bf16) * B(32x16 bf16) + C(16x16 f32)
  return __builtin_amdgcn_wmma_f32_16x16x32_bf16(
      false, a, false, b, (short)0, c, false, false);
}

// ---------------------------------------------------------------------------
// Phase 0a: init h ping-pong buffer from h0 and zero the grid-sync counter.
// ---------------------------------------------------------------------------
__global__ void lstm_init_kernel(const float* __restrict__ h0,
                                 float* __restrict__ hbuf,
                                 unsigned int* __restrict__ gsync) {
  int i = blockIdx.x * blockDim.x + threadIdx.x;
  if (i < BATCH * HID) hbuf[i] = h0[i];       // hbuf[buffer 0]
  if (i == 0) *gsync = 0u;
}

// ---------------------------------------------------------------------------
// Phase 0b: pack a [256 x 1024] fp32 weight into bf16 WMMA B-operand tiles.
// Tile (kt,nt) covers K rows kt*32..+31, N cols nt*16..+15.
// Packed layout: P[(nt*8+kt)*512 + lane*16 + j]  (32 lanes x 16 bf16 = 1 KB).
// Lane L: N = nt*16 + (L&15), K = kt*32 + (L>>4)*16 + j.
// ---------------------------------------------------------------------------
__global__ void pack_weight_kernel(const float* __restrict__ W,
                                   bf16_t* __restrict__ P) {
  int gid  = blockIdx.x * blockDim.x + threadIdx.x;   // 16384 threads
  int lane = gid & 31;
  int tile = gid >> 5;                                // 0..511 = nt*8 + kt
  if (tile >= 512) return;
  int kt = tile & 7, nt = tile >> 3;
  int col = nt * 16 + (lane & 15);
  int k0  = kt * 32 + (lane >> 4) * 16;
  bf16_t* dst = P + (size_t)tile * 512 + lane * 16;
#pragma unroll
  for (int j = 0; j < 16; ++j)
    dst[j] = (bf16_t)W[(size_t)(k0 + j) * G4H + col];
}

// ---------------------------------------------------------------------------
// Phase 1: x_proj = xs @ Wx   (131072 x 256) * (256 x 1024) -> f16 out.
// Block = 256 threads (8 waves) covers M-block of 32 rows x all 1024 cols.
// A (xs) staged once in LDS as bf16 A-tiles; B read from packed Wx (L2-hot).
// ---------------------------------------------------------------------------
__global__ __launch_bounds__(256) void xproj_gemm_kernel(
    const float* __restrict__ xs, const bf16_t* __restrict__ Wxp,
    _Float16* __restrict__ xproj) {
  __shared__ bf16_t As[16][512];   // 16 A-tiles (2 Mtiles x 8 Ktiles), 16 KB

  const int tid  = threadIdx.x;
  const int lane = tid & 31;
  const int wave = tid >> 5;
  const int m0   = blockIdx.x * 32;

  // Stage A tiles: slot s = (tile t, lane l); t = kt*2 + mt.
  for (int s = tid; s < 512; s += 256) {
    int t = s >> 5, l = s & 31;
    int mt = t & 1, kt = t >> 1;
    int row = m0 + mt * 16 + (l & 15);
    int k0  = kt * 32 + (l >> 4) * 16;
    const float* src = xs + (size_t)row * HID + k0;
    bf16_t* d = &As[t][l * 16];
#pragma unroll
    for (int j = 0; j < 16; ++j) d[j] = (bf16_t)src[j];
  }
  __syncthreads();

  // Wave w handles N-tiles w*8 .. w*8+7.
  for (int n = 0; n < 8; ++n) {
    int nt = wave * 8 + n;
    v8f c0 = {}, c1 = {};
#pragma unroll
    for (int kt = 0; kt < 8; ++kt) {
      v16bf bt = *(const v16bf*)(Wxp + ((size_t)(nt * 8 + kt)) * 512 + lane * 16);
      v16bf a0 = *(const v16bf*)(&As[kt * 2 + 0][lane * 16]);
      v16bf a1 = *(const v16bf*)(&As[kt * 2 + 1][lane * 16]);
      c0 = wmma_bf16(a0, bt, c0);
      c1 = wmma_bf16(a1, bt, c1);
    }
    // C layout: col = nt*16 + (lane&15); rows (lane>>4)*8 + r (+16 for c1).
    int col   = nt * 16 + (lane & 15);
    int rbase = m0 + (lane >> 4) * 8;
#pragma unroll
    for (int r = 0; r < 8; ++r) {
      xproj[(size_t)(rbase + r) * G4H + col]      = (_Float16)c0[r];
      xproj[(size_t)(rbase + r + 16) * G4H + col] = (_Float16)c1[r];
    }
  }
}

// ---------------------------------------------------------------------------
// Phase 2: persistent LSTM scan.  8 WGs x 256 threads.  WG k owns hidden
// units k*32..k*32+31 (gate cols g*256 + k*32 + j).  Wh slice lives in VGPRs.
// Per step: TDM-prefetch x_proj slice -> LDS; stage h -> bf16 A-tiles in LDS;
// WMMA h@Wh; gates + cell update in registers; grid barrier.
// ---------------------------------------------------------------------------
__global__ __launch_bounds__(256) void lstm_scan_kernel(
    const float* __restrict__ c0, const _Float16* __restrict__ xproj,
    const bf16_t* __restrict__ Whp, const float* __restrict__ bias_g,
    float* __restrict__ ys, float* __restrict__ hbuf,
    unsigned int* __restrict__ gsync) {
  __shared__ _Float16 xp[BATCH * 4 * 32];  // 8 KB  [batch][gate][hid32]
  __shared__ bf16_t   Ah[16][512];         // 16 KB h as A-operand tiles
  __shared__ float    zb[128][32];         // 16 KB [local col][batch]

  const int tid  = threadIdx.x;
  const int lane = tid & 31;
  const int wave = tid >> 5;
  const int wg   = blockIdx.x;

  // Wave's B-operand slice of Wh (8 K-tiles of one N-tile) -> 64 VGPRs/lane.
  const int ntw = (wave >> 1) * 16 + wg * 2 + (wave & 1);
  v16bf bw[8];
#pragma unroll
  for (int kt = 0; kt < 8; ++kt)
    bw[kt] = *(const v16bf*)(Whp + ((size_t)(ntw * 8 + kt)) * 512 + lane * 16);

  // Per-thread state: hidden col = wg*32 + (tid&31); batches (tid>>5)+8r.
  const int hid  = tid & 31;
  const int hcol = wg * 32 + hid;
  float bsc[4], creg[4];
#pragma unroll
  for (int g = 0; g < 4; ++g) bsc[g] = bias_g[g * 256 + hcol];
#pragma unroll
  for (int r = 0; r < 4; ++r)
    creg[r] = c0[(size_t)((tid >> 5) + 8 * r) * HID + hcol];

  const unsigned xp_lds = (unsigned)(size_t)(void*)xp;

  for (int t = 0; t < T_STEPS; ++t) {
    // ---- 1. Async-prefetch this step's x_proj slice with the TDM -------
    // 3-D tile: dim0 = 32 f16 (contiguous cols), dim1 = 4 gates (stride 256),
    // dim2 = 32 batch rows (stride 1024).  Units = data_size (2 B).
#if __has_builtin(__builtin_amdgcn_tensor_load_to_lds)
    if (wave == 0) {
      size_t gaddr = (size_t)(const void*)(xproj +
                     ((size_t)t * BATCH * G4H + wg * 32));
      unsigned glo = (unsigned)gaddr, ghi = (unsigned)(gaddr >> 32);
      v4u g0 = { 1u,                                   // count=1, user D#
                 xp_lds,                               // LDS dest
                 glo,
                 (ghi & 0x01FFFFFFu) | (2u << 30) };   // addr[56:32] | type=2
      const unsigned td0 = 0x100000u, td1 = 0x100000u; // huge dims: no OOB clip
      v8i g1 = {};
      g1[0] = (int)(1u << 16);                         // data_size=1 (2 bytes)
      g1[1] = (int)((td0 & 0xFFFFu) << 16);            // tensor_dim0 lo16
      g1[2] = (int)(((td0 >> 16) & 0xFFFFu) | ((td1 & 0xFFFFu) << 16));
      g1[3] = (int)(((td1 >> 16) & 0xFFFFu) | (32u << 16)); // tile_dim0=32
      g1[4] = (int)(4u | (32u << 16));                 // tile_dim1=4, tile_dim2=32
      g1[5] = (int)256;                                // dim0_stride = 256 elems
      g1[6] = (int)(1024u << 16);                      // dim1_stride lo16 = 1024
      g1[7] = 0;                                       // dim1_stride hi
      v4i g2 = { (int)0x100000, 0, 0, 0 };             // tensor_dim2 big
      v4i g3 = { 0, 0, 0, 0 };
#if defined(__clang_major__) && __clang_major__ >= 23
      v8i g5 = {};
      __builtin_amdgcn_tensor_load_to_lds(g0, g1, g2, g3, g5, 0);
#else
      __builtin_amdgcn_tensor_load_to_lds(g0, g1, g2, g3, 0);
#endif
    }
#else
    // Fallback: cooperative copy (no TDM builtin on this toolchain).
    {
      const _Float16* src = xproj + ((size_t)t * BATCH * G4H + wg * 32);
      for (int s = tid; s < BATCH * 4 * 32; s += 256) {
        int h32 = s & 31, g = (s >> 5) & 3, bq = s >> 7;
        xp[s] = src[(size_t)bq * G4H + g * 256 + h32];
      }
    }
#endif

    // ---- 2. Stage h (fp32, L2) -> bf16 A-operand tiles in LDS ----------
    const float* hsrc = hbuf + (size_t)(t & 1) * (BATCH * HID);
    for (int s = tid; s < 512; s += 256) {
      int tt = s >> 5, l = s & 31;
      int mt = tt & 1, kt = tt >> 1;
      int row = mt * 16 + (l & 15);
      int k0  = kt * 32 + (l >> 4) * 16;
      const float* src = hsrc + (size_t)row * HID + k0;
      bf16_t* d = &Ah[tt][l * 16];
#pragma unroll
      for (int j = 0; j < 16; ++j) d[j] = (bf16_t)src[j];
    }
    __syncthreads();

    // ---- 3. z_h = h @ Wh slice via WMMA (K=256 in 8 steps) -------------
    v8f z0 = {}, z1 = {};
#pragma unroll
    for (int kt = 0; kt < 8; ++kt) {
      v16bf a0 = *(const v16bf*)(&Ah[kt * 2 + 0][lane * 16]);
      v16bf a1 = *(const v16bf*)(&Ah[kt * 2 + 1][lane * 16]);
      z0 = wmma_bf16(a0, bw[kt], z0);
      z1 = wmma_bf16(a1, bw[kt], z1);
    }
    // Spill C tiles to LDS for the cross-wave gate stage.
    {
      int c_loc = wave * 16 + (lane & 15);
      int mb    = (lane >> 4) * 8;
#pragma unroll
      for (int r = 0; r < 8; ++r) {
        zb[c_loc][mb + r]      = z0[r];
        zb[c_loc][mb + r + 16] = z1[r];
      }
    }
    if (wave == 0) __builtin_amdgcn_s_wait_tensorcnt(0);
    __syncthreads();

    // ---- 4. Gates + cell update (c in registers) -----------------------
    float* hdst = hbuf + (size_t)((t + 1) & 1) * (BATCH * HID);
#pragma unroll
    for (int r = 0; r < 4; ++r) {
      int bq = (tid >> 5) + 8 * r;
      float zi = zb[0 * 32 + hid][bq] + (float)xp[(bq * 4 + 0) * 32 + hid] + bsc[0];
      float zf = zb[1 * 32 + hid][bq] + (float)xp[(bq * 4 + 1) * 32 + hid] + bsc[1];
      float zg = zb[2 * 32 + hid][bq] + (float)xp[(bq * 4 + 2) * 32 + hid] + bsc[2];
      float zo = zb[3 * 32 + hid][bq] + (float)xp[(bq * 4 + 3) * 32 + hid] + bsc[3];
      float ig = 1.f / (1.f + __expf(-zi));
      float fg = 1.f / (1.f + __expf(-zf));
      float og = 1.f / (1.f + __expf(-zo));
      float gg = tanhf(zg);
      float c  = fg * creg[r] + ig * gg;
      creg[r]  = c;
      float h  = og * tanhf(c);
      ys[((size_t)t * BATCH + bq) * HID + hcol] = h;
      hdst[(size_t)bq * HID + hcol] = h;
    }

    // ---- 5. Grid barrier (monotonic counter, release/acquire) ----------
    __threadfence();
    __syncthreads();
    if (tid == 0) {
      __hip_atomic_fetch_add(gsync, 1u, __ATOMIC_RELEASE,
                             __HIP_MEMORY_SCOPE_AGENT);
      unsigned target = (unsigned)(t + 1) * NWG;
      while (__hip_atomic_load(gsync, __ATOMIC_ACQUIRE,
                               __HIP_MEMORY_SCOPE_AGENT) < target)
        __builtin_amdgcn_s_sleep(1);
    }
    __syncthreads();
  }
}

// ---------------------------------------------------------------------------
// Host-side launcher.
// Inputs: 0=c0[B,H] 1=h0[B,H] 2=xs[T,B,H] 3=Wx[H,4H] 4=Wh[H,4H] 5=b[4H] (f32).
// Output: ys[T,B,H] f32.
// ---------------------------------------------------------------------------
extern "C" void kernel_launch(void* const* d_in, const int* in_sizes, int n_in,
                              void* d_out, int out_size, void* d_ws,
                              size_t ws_size, hipStream_t stream) {
  const float* c0 = (const float*)d_in[0];
  const float* h0 = (const float*)d_in[1];
  const float* xs = (const float*)d_in[2];
  const float* Wx = (const float*)d_in[3];
  const float* Wh = (const float*)d_in[4];
  const float* bg = (const float*)d_in[5];
  float* ys = (float*)d_out;

  // Workspace layout.
  const size_t xproj_bytes = (size_t)TBROWS * G4H * sizeof(_Float16); // 268 MB
  const size_t packw_bytes = (size_t)512 * 512 * sizeof(bf16_t);      // 512 KB
  char* ws = (char*)d_ws;
  _Float16* xproj = (_Float16*)ws;
  bf16_t*   Wxp   = (bf16_t*)(ws + xproj_bytes);
  bf16_t*   Whp   = (bf16_t*)(ws + xproj_bytes + packw_bytes);
  float*    hbuf  = (float*)(ws + xproj_bytes + 2 * packw_bytes);     // 2x32x256
  unsigned* gsync = (unsigned*)(ws + xproj_bytes + 2 * packw_bytes +
                                2 * BATCH * HID * sizeof(float));

  // Phase 0: init + pack weights (fp32 -> bf16 WMMA operand layout).
  lstm_init_kernel<<<(BATCH * HID + 255) / 256, 256, 0, stream>>>(h0, hbuf, gsync);
  pack_weight_kernel<<<64, 256, 0, stream>>>(Wx, Wxp);
  pack_weight_kernel<<<64, 256, 0, stream>>>(Wh, Whp);

  // Phase 1: big memory-bound GEMM for x_proj (bf16 WMMA, f32 accumulate).
  xproj_gemm_kernel<<<TBROWS / 32, 256, 0, stream>>>(xs, Wxp, xproj);

  // Phase 2: persistent sequential LSTM scan.
  lstm_scan_kernel<<<NWG, 256, 0, stream>>>(c0, xproj, Whp, bg, ys, hbuf, gsync);
}
